// CrossAttention_64742337020415
// MI455X (gfx1250) — compile-verified, matrix operand order checked by reading, not run on hardware
//
#include <hip/hip_runtime.h>
#include <stddef.h>
#include <stdint.h>

// ---------------------------------------------------------------------------
// CDNA5 (gfx1250) mixed-precision implementation of the style-injected
// cross-attention reference.  All matmuls run on v_wmma_f32_16x16x32_bf16
// (wave32, 16x16 tiles).  The dominant attn@[V,V^2] pass computes a full
// 16x80 row panel per wave so the 268 MB attention matrix is streamed from
// HBM exactly once.  The fused lambda/top-k/softmax kernel stages its row
// into LDS with global_load_async_to_lds_b128 (ASYNCcnt-tracked).
// ---------------------------------------------------------------------------

typedef __bf16 bf16_t;
typedef __attribute__((ext_vector_type(16))) __bf16 v16bf;
typedef __attribute__((ext_vector_type(8)))  __bf16 v8bf;
typedef __attribute__((ext_vector_type(8)))  float  v8f;

static constexpr int H     = 8;
static constexpr int NQ    = 2048;
static constexpr int NS    = 2048;
static constexpr int DH    = 80;
static constexpr int DP    = 96;            // head dim padded to multiple of 32
static constexpr int QD    = 640;           // H * DH
static constexpr int NTOK  = NS + NQ;       // 4096
static constexpr int KTOP  = 128;
static constexpr int NT_DH = DH / 16;       // 5 n-tiles in the AV pass
static constexpr int NSTRIP = 8;            // n-tiles per wave in logit pass
static constexpr float T_INJ   = 1.2f;
static constexpr float GAMMA_Q = 0.3f;
static constexpr float LAM_MAX = 2.5f;
static constexpr float SCALE   = 0.11180339887498949f;   // 80^-0.5
static constexpr float NEG_BIG = -3.4e38f;

// ----------------------------- WMMA fragments ------------------------------
// 16-bit A-matrix 16x32 layout (ISA 7.12.2): lane L holds row m = L&15;
// half = L>>4; element e<8 -> K = 8*half + e ; e>=8 -> K = 16 + 8*half + e-8.
// B (NxK operand) uses the identical mapping with n = L&15 as the row index.

__device__ __forceinline__ v16bf frag_bf16(const bf16_t* __restrict__ p,
                                           int ld, int m0, int k0, int lane) {
  int m  = m0 + (lane & 15);
  int kb = k0 + ((lane >> 4) << 3);
  const bf16_t* row = p + (size_t)m * ld + kb;
  v8bf lo = *(const v8bf*)(row);        // K = kb .. kb+7
  v8bf hi = *(const v8bf*)(row + 16);   // K = kb+16 .. kb+23
  return __builtin_shufflevector(lo, hi, 0, 1, 2, 3, 4, 5, 6, 7,
                                         8, 9, 10, 11, 12, 13, 14, 15);
}

__device__ __forceinline__ v16bf frag_f32(const float* __restrict__ p,
                                          int ld, int m0, int k0, int lane) {
  int m  = m0 + (lane & 15);
  int kb = k0 + ((lane >> 4) << 3);
  const float* row = p + (size_t)m * ld + kb;
  float4 a0 = *(const float4*)(row);
  float4 a1 = *(const float4*)(row + 4);
  float4 b0 = *(const float4*)(row + 16);
  float4 b1 = *(const float4*)(row + 20);
  v16bf f;
  f[0]  = (__bf16)a0.x; f[1]  = (__bf16)a0.y; f[2]  = (__bf16)a0.z; f[3]  = (__bf16)a0.w;
  f[4]  = (__bf16)a1.x; f[5]  = (__bf16)a1.y; f[6]  = (__bf16)a1.z; f[7]  = (__bf16)a1.w;
  f[8]  = (__bf16)b0.x; f[9]  = (__bf16)b0.y; f[10] = (__bf16)b0.z; f[11] = (__bf16)b0.w;
  f[12] = (__bf16)b1.x; f[13] = (__bf16)b1.y; f[14] = (__bf16)b1.z; f[15] = (__bf16)b1.w;
  return f;
}

#define WMMA_BF16(A, B, C)                                                    \
  __builtin_amdgcn_wmma_f32_16x16x32_bf16(false, (A), false, (B), (short)0,   \
                                          (C), false, false)

// C/D layout: lane L covers n = L&15; VGPR r holds m = r + 8*(L>>4).

// ------------------- generic C = A(MxK) * B(NxK)^T GEMM --------------------
// Used for the three input projections and the output projection.
__global__ __launch_bounds__(256) void gemm_bf16_nt(
    const bf16_t* __restrict__ A, const bf16_t* __restrict__ B,
    float* __restrict__ C, int M, int N, int K,
    int lda, int ldb, int ldc, float scale, const float* __restrict__ bias) {
  int lane = threadIdx.x & 31;
  int wave = threadIdx.x >> 5;
  int tiles_n = N >> 4;
  int tile = blockIdx.x * 8 + wave;
  if (tile >= (M >> 4) * tiles_n) return;       // wave-uniform (EXEC stays full)
  int tm = (tile / tiles_n) << 4;
  int tn = (tile % tiles_n) << 4;
  v8f acc = {0.f, 0.f, 0.f, 0.f, 0.f, 0.f, 0.f, 0.f};
  for (int k0 = 0; k0 < K; k0 += 32) {
    if (k0 + 64 < K) {                          // software prefetch of next slab
      __builtin_prefetch(A + (size_t)(tm + (lane & 15)) * lda + k0 + 64);
      __builtin_prefetch(B + (size_t)(tn + (lane & 15)) * ldb + k0 + 64);
    }
    v16bf a = frag_bf16(A, lda, tm, k0, lane);
    v16bf b = frag_bf16(B, ldb, tn, k0, lane);
    acc = WMMA_BF16(a, b, acc);
  }
  int n  = tn + (lane & 15);
  int mb = tm + ((lane >> 4) << 3);
  float bv = bias ? bias[n] : 0.0f;
#pragma unroll
  for (int r = 0; r < 8; ++r)
    C[(size_t)(mb + r) * ldc + n] = acc[r] * scale + bv;
}

// ---- logits: per-head C(2048x2048) = A(2048xDP) * B(2048xDP)^T ------------
// Each wave computes an 8-tile N-strip: A fragments loaded once per K-step,
// amortized over 8 WMMAs.  coff selects style (0) / content (NS) columns.
__global__ __launch_bounds__(256) void gemm_logits(
    const bf16_t* __restrict__ A, const bf16_t* __restrict__ B,
    float* __restrict__ C, int coff, float scale) {
  int lane = threadIdx.x & 31;
  int wave = threadIdx.x >> 5;
  int h = blockIdx.y;
  const bf16_t* Ab = A + (long long)h * NQ * DP;
  const bf16_t* Bb = B + (long long)h * NS * DP;
  float* Cb = C + (long long)h * NQ * NTOK + coff;
  const int strips_n = NS / (16 * NSTRIP);       // 16 strips across N
  int sid = blockIdx.x * 8 + wave;               // 0 .. 2047
  int tm  = (sid / strips_n) << 4;
  int tn0 = (sid % strips_n) * (16 * NSTRIP);
  v8f z = {0.f, 0.f, 0.f, 0.f, 0.f, 0.f, 0.f, 0.f};
  v8f acc[NSTRIP];
#pragma unroll
  for (int i = 0; i < NSTRIP; ++i) acc[i] = z;
#pragma unroll
  for (int k0 = 0; k0 < DP; k0 += 32) {
    v16bf a = frag_bf16(Ab, DP, tm, k0, lane);
#pragma unroll
    for (int nt = 0; nt < NSTRIP; ++nt) {
      v16bf b = frag_bf16(Bb, DP, tn0 + nt * 16, k0, lane);
      acc[nt] = WMMA_BF16(a, b, acc[nt]);
    }
  }
  int mb = tm + ((lane >> 4) << 3);
  int nl = lane & 15;
#pragma unroll
  for (int nt = 0; nt < NSTRIP; ++nt) {
    int n = tn0 + nt * 16 + nl;
#pragma unroll
    for (int r = 0; r < 8; ++r)
      Cb[(size_t)(mb + r) * NTOK + n] = acc[nt][r] * scale;
  }
}

// ---- dual AV GEMM: Vmean = attn*V, Vsq = attn*(V*V) -----------------------
// One 16x80 row panel per wave: the f32 attn row (cvt to bf16 in registers)
// is read from HBM exactly once; 10 WMMAs per K-step.
__global__ __launch_bounds__(256) void gemm_attn_v(
    const float* __restrict__ A, const bf16_t* __restrict__ B0,
    const bf16_t* __restrict__ B1, float* __restrict__ C0,
    float* __restrict__ C1) {
  int lane = threadIdx.x & 31;
  int wave = threadIdx.x >> 5;
  int h = blockIdx.y;
  const float*  Ab  = A  + (long long)h * NQ * NTOK;
  const bf16_t* B0b = B0 + (long long)h * DH * NTOK;
  const bf16_t* B1b = B1 + (long long)h * DH * NTOK;
  float* C0b = C0 + (long long)h * NQ * DH;
  float* C1b = C1 + (long long)h * NQ * DH;
  int tm = (blockIdx.x * 8 + wave) << 4;         // 128 row tiles, grid.x = 16
  v8f z = {0.f, 0.f, 0.f, 0.f, 0.f, 0.f, 0.f, 0.f};
  v8f acc0[NT_DH], acc1[NT_DH];
#pragma unroll
  for (int i = 0; i < NT_DH; ++i) { acc0[i] = z; acc1[i] = z; }
  for (int k0 = 0; k0 < NTOK; k0 += 32) {
    if (k0 + 64 < NTOK)
      __builtin_prefetch(Ab + (size_t)(tm + (lane & 15)) * NTOK + k0 + 64);
    v16bf a = frag_f32(Ab, NTOK, tm, k0, lane);
#pragma unroll
    for (int nt = 0; nt < NT_DH; ++nt) {
      v16bf b0 = frag_bf16(B0b, NTOK, nt * 16, k0, lane);
      acc0[nt] = WMMA_BF16(a, b0, acc0[nt]);
      v16bf b1 = frag_bf16(B1b, NTOK, nt * 16, k0, lane);
      acc1[nt] = WMMA_BF16(a, b1, acc1[nt]);
    }
  }
  int mb = tm + ((lane >> 4) << 3);
  int nl = lane & 15;
#pragma unroll
  for (int nt = 0; nt < NT_DH; ++nt) {
#pragma unroll
    for (int r = 0; r < 8; ++r) {
      C0b[(size_t)(mb + r) * DH + nt * 16 + nl] = acc0[nt][r];
      C1b[(size_t)(mb + r) * DH + nt * 16 + nl] = acc1[nt][r];
    }
  }
}

// --------------------------- block reductions ------------------------------
__device__ __forceinline__ float blk_max(float v, float* red) {
  int t = threadIdx.x;
  red[t] = v; __syncthreads();
#pragma unroll
  for (int s = 128; s >= 1; s >>= 1) {
    if (t < s) red[t] = fmaxf(red[t], red[t + s]);
    __syncthreads();
  }
  float r = red[0]; __syncthreads(); return r;
}
__device__ __forceinline__ float blk_min(float v, float* red) {
  int t = threadIdx.x;
  red[t] = v; __syncthreads();
#pragma unroll
  for (int s = 128; s >= 1; s >>= 1) {
    if (t < s) red[t] = fminf(red[t], red[t + s]);
    __syncthreads();
  }
  float r = red[0]; __syncthreads(); return r;
}
__device__ __forceinline__ float blk_sum(float v, float* red) {
  int t = threadIdx.x;
  red[t] = v; __syncthreads();
#pragma unroll
  for (int s = 128; s >= 1; s >>= 1) {
    if (t < s) red[t] += red[t + s];
    __syncthreads();
  }
  float r = red[0]; __syncthreads(); return r;
}

// ------------------------- elementwise / pack kernels ----------------------
__global__ void f32_to_bf16(const float* __restrict__ s, bf16_t* __restrict__ d, int n) {
  int i = blockIdx.x * blockDim.x + threadIdx.x;
  if (i < n) d[i] = (bf16_t)s[i];
}

// q = GAMMA*Qc + (1-GAMMA)*q_cs; keep f32 copy + zero-padded bf16 (H,NQ,DP)
__global__ void pack_q(const float* __restrict__ qcs, const float* __restrict__ Qc,
                       float* __restrict__ qf, bf16_t* __restrict__ qb) {
  int idx = blockIdx.x * blockDim.x + threadIdx.x;
  if (idx >= H * NQ * DP) return;
  int h = idx / (NQ * DP);
  int rem = idx - h * NQ * DP;
  int n = rem / DP, d = rem - n * DP;
  float v = 0.f;
  if (d < DH) {
    v = GAMMA_Q * Qc[((size_t)h * NQ + n) * DH + d] +
        (1.f - GAMMA_Q) * qcs[(size_t)n * QD + h * DH + d];
    qf[((size_t)h * NQ + n) * DH + d] = v;
  }
  qb[idx] = (bf16_t)v;
}

// mode 0: src (H,N,DH) per-head ; mode 1: src (N, H*DH) interleaved
__global__ void pack_head_bf16(const float* __restrict__ src, bf16_t* __restrict__ dst,
                               int mode) {
  int idx = blockIdx.x * blockDim.x + threadIdx.x;
  if (idx >= H * NS * DP) return;
  int h = idx / (NS * DP);
  int rem = idx - h * NS * DP;
  int n = rem / DP, d = rem - n * DP;
  float v = 0.f;
  if (d < DH)
    v = (mode == 0) ? src[((size_t)h * NS + n) * DH + d]
                    : src[(size_t)n * QD + h * DH + d];
  dst[idx] = (bf16_t)v;
}

// transposed concat V: vt(h,d,j) = [V_s ; v_cs], vsqt = square
__global__ void pack_vt(const float* __restrict__ Vsrc, const float* __restrict__ vcs,
                        bf16_t* __restrict__ vt, bf16_t* __restrict__ vsqt) {
  int idx = blockIdx.x * blockDim.x + threadIdx.x;
  if (idx >= H * DH * NTOK) return;
  int h = idx / (DH * NTOK);
  int rem = idx - h * DH * NTOK;
  int d = rem / NTOK, j = rem - d * NTOK;
  float v = (j < NS) ? Vsrc[((size_t)h * NS + j) * DH + d]
                     : vcs[(size_t)(j - NS) * QD + h * DH + d];
  vt[idx]   = (bf16_t)v;
  vsqt[idx] = (bf16_t)(v * v);
}

// rbuf(h,n) = lse(style row) - lse(content row)   (pre-lambda logits)
__global__ __launch_bounds__(256) void row_lse(const float* __restrict__ logits,
                                               float* __restrict__ rbuf) {
  __shared__ float red[256];
  int row = blockIdx.x;                  // h*NQ + n
  const float* p = logits + (size_t)row * NTOK;
  int t = threadIdx.x;
  float ms = NEG_BIG, mc = NEG_BIG;
  for (int i = t; i < NS; i += 256) ms = fmaxf(ms, p[i]);
  ms = blk_max(ms, red);
  for (int i = t; i < NQ; i += 256) mc = fmaxf(mc, p[NS + i]);
  mc = blk_max(mc, red);
  float ss = 0.f, sc = 0.f;
  for (int i = t; i < NS; i += 256) ss += __expf(p[i] - ms);
  ss = blk_sum(ss, red);
  for (int i = t; i < NQ; i += 256) sc += __expf(p[NS + i] - mc);
  sc = blk_sum(sc, red);
  if (t == 0) rbuf[row] = (ms + __logf(ss)) - (mc + __logf(sc));
}

__global__ void compute_lam(const float* __restrict__ rbuf, float* __restrict__ lam) {
  int n = blockIdx.x * blockDim.x + threadIdx.x;
  if (n >= NQ) return;
  float s = 0.f;
  for (int h = 0; h < H; ++h)
    s += fminf(fmaxf(rbuf[h * NQ + n], -20.f), 20.f);
  float r = s * (1.0f / H);
  lam[n] = 1.0f + (LAM_MAX - 1.0f) * (1.0f / (1.0f + __expf(-r)));
}

// Async memory->LDS copy of 16 bytes per lane (CDNA5 ASYNCcnt-tracked path).
// LDS byte offset = low 32 bits of the generic shared-pointer (ISA 10.2).
__device__ __forceinline__ void async_to_lds16(void* lds_ptr, const void* gptr) {
  uint32_t dsa = (uint32_t)(uintptr_t)lds_ptr;
  unsigned long long ga = (unsigned long long)(uintptr_t)gptr;
  asm volatile("global_load_async_to_lds_b128 %0, %1, off"
               :: "v"(dsa), "v"(ga) : "memory");
}
__device__ __forceinline__ void wait_async_lds() {
  asm volatile("s_wait_asynccnt 0x0" ::: "memory");
}

// lambda-scale style, top-k(128) threshold by bisection, fused softmax over
// [style(masked) , content]; writes attn probs in place.
__global__ __launch_bounds__(256) void topk_softmax(float* __restrict__ logits,
                                                    const float* __restrict__ lam) {
  __shared__ float s_sty[NS];
  __shared__ float s_con[NQ];
  __shared__ float red[256];
  int row = blockIdx.x;
  int n = row & (NQ - 1);
  float* p = logits + (size_t)row * NTOK;
  int t = threadIdx.x;
  // Stage the whole 16 KB row into LDS: 4 async b128 copies per thread.
#pragma unroll
  for (int r = 0; r < 2; ++r) {
    int e = (r * 256 + t) * 4;                  // float index, 16B aligned
    async_to_lds16(&s_sty[e], p + e);
    async_to_lds16(&s_con[e], p + NS + e);
  }
  float lv = lam[n];
  wait_async_lds();
  __syncthreads();
  float mx = NEG_BIG, mn = -NEG_BIG;
  for (int i = t; i < NS; i += 256) {           // lambda-scale in LDS
    float v = s_sty[i] * lv;
    s_sty[i] = v;
    mx = fmaxf(mx, v);
    mn = fminf(mn, v);
  }
  float hi = blk_max(mx, red);
  float lo = blk_min(mn, red);
  float smax = hi;
  // largest thr with count(style >= thr) >= KTOP  (ties keep >= KTOP, as topk)
  for (int it = 0; it < 32; ++it) {
    float mid = 0.5f * (lo + hi);
    float c = 0.f;
    for (int i = t; i < NS; i += 256) c += (s_sty[i] >= mid) ? 1.f : 0.f;
    c = blk_sum(c, red);
    if (c >= (float)KTOP) lo = mid; else hi = mid;
  }
  float thr = lo;
  float cm = NEG_BIG;
  for (int i = t; i < NQ; i += 256) cm = fmaxf(cm, s_con[i]);
  cm = blk_max(cm, red);
  float m = fmaxf(smax, cm);          // style max always survives top-k
  float s = 0.f;
  for (int i = t; i < NS; i += 256) {
    float v = s_sty[i];
    float e = (v >= thr) ? __expf(v - m) : 0.f;
    s_sty[i] = e;
    s += e;
  }
  for (int i = t; i < NQ; i += 256) {
    float e = __expf(s_con[i] - m);
    s_con[i] = e;
    s += e;
  }
  float Z = blk_sum(s, red);
  float rz = 1.0f / Z;
  for (int i = t; i < NS; i += 256) p[i] = s_sty[i] * rz;
  for (int i = t; i < NQ; i += 256) p[NS + i] = s_con[i] * rz;
}

// per-(h,d) mean / rsqrt(var+eps) of q over the token axis
__global__ __launch_bounds__(256) void q_stats(const float* __restrict__ qf,
                                               float* __restrict__ mu,
                                               float* __restrict__ inv) {
  __shared__ float red[256];
  int hd = blockIdx.x;                 // 0 .. H*DH-1
  int h = hd / DH, d = hd - h * DH;
  const float* base = qf + (size_t)h * NQ * DH + d;
  int t = threadIdx.x;
  float s = 0.f, ss = 0.f;
  for (int nn = t; nn < NQ; nn += 256) {
    float v = base[(size_t)nn * DH];
    s += v; ss += v * v;
  }
  s = blk_sum(s, red);
  ss = blk_sum(ss, red);
  if (t == 0) {
    float m = s * (1.0f / NQ);
    float var = ss * (1.0f / NQ) - m * m;
    mu[hd] = m;
    inv[hd] = rsqrtf(var + 1e-5f);
  }
}

// out = Vmean + tanh(g) * Vstd * Qn   (out == Vmean, so residual simplifies)
__global__ void modulate(const float* __restrict__ Vm, const float* __restrict__ Vq,
                         const float* __restrict__ qf, const float* __restrict__ mu,
                         const float* __restrict__ inv, const float* __restrict__ gate,
                         bf16_t* __restrict__ y) {
  int idx = blockIdx.x * blockDim.x + threadIdx.x;
  if (idx >= H * NQ * DH) return;
  float g = tanhf(gate[0]);
  int h = idx / (NQ * DH);
  int rem = idx - h * NQ * DH;
  int n = rem / DH, d = rem - n * DH;
  float vm = Vm[idx], vq = Vq[idx];
  float stdv = sqrtf(fmaxf(vq - vm * vm, 1e-5f));
  int hd = h * DH + d;
  float qn = (qf[idx] - mu[hd]) * inv[hd];
  float val = vm + g * (stdv * qn);
  y[(size_t)n * QD + hd] = (bf16_t)val;   // (n, h*DH+d) layout for out-proj
}

// ------------------------------- launcher ----------------------------------
extern "C" void kernel_launch(void* const* d_in, const int* in_sizes, int n_in,
                              void* d_out, int out_size, void* d_ws, size_t ws_size,
                              hipStream_t stream) {
  (void)in_sizes; (void)n_in; (void)out_size; (void)ws_size;
  const float* x    = (const float*)d_in[0];   // (1,2048,640)
  const float* Qc   = (const float*)d_in[1];   // (8,2048,80)
  const float* Ks   = (const float*)d_in[2];   // (8,2048,80)
  const float* Vs   = (const float*)d_in[3];   // (8,2048,80)
  const float* Wq   = (const float*)d_in[4];   // (640,640)
  const float* Wk   = (const float*)d_in[5];
  const float* Wv   = (const float*)d_in[6];
  const float* Wo   = (const float*)d_in[7];
  const float* bo   = (const float*)d_in[8];   // (640)
  const float* gate = (const float*)d_in[9];   // (1)
  float* out = (float*)d_out;                  // (1,2048,640) f32

  char* w = (char*)d_ws;
  auto carve = [&](size_t bytes) -> char* {
    char* p = w;
    w += (bytes + 255) & ~(size_t)255;
    return p;
  };
  bf16_t* xb    = (bf16_t*)carve((size_t)NQ * QD * sizeof(bf16_t));
  bf16_t* Wqb   = (bf16_t*)carve((size_t)QD * QD * sizeof(bf16_t));
  bf16_t* Wkb   = (bf16_t*)carve((size_t)QD * QD * sizeof(bf16_t));
  bf16_t* Wvb   = (bf16_t*)carve((size_t)QD * QD * sizeof(bf16_t));
  bf16_t* Wob   = (bf16_t*)carve((size_t)QD * QD * sizeof(bf16_t));
  float*  qcs   = (float*)carve((size_t)NQ * QD * sizeof(float));
  float*  kcs   = (float*)carve((size_t)NQ * QD * sizeof(float));
  float*  vcs   = (float*)carve((size_t)NQ * QD * sizeof(float));
  float*  qf    = (float*)carve((size_t)H * NQ * DH * sizeof(float));
  bf16_t* qb    = (bf16_t*)carve((size_t)H * NQ * DP * sizeof(bf16_t));
  bf16_t* Ksb   = (bf16_t*)carve((size_t)H * NS * DP * sizeof(bf16_t));
  bf16_t* kcsb  = (bf16_t*)carve((size_t)H * NS * DP * sizeof(bf16_t));
  bf16_t* vtb   = (bf16_t*)carve((size_t)H * DH * NTOK * sizeof(bf16_t));
  bf16_t* vsqtb = (bf16_t*)carve((size_t)H * DH * NTOK * sizeof(bf16_t));
  float*  logit = (float*)carve((size_t)H * NQ * NTOK * sizeof(float)); // 268 MB
  float*  rbuf  = (float*)carve((size_t)H * NQ * sizeof(float));
  float*  lamb  = (float*)carve((size_t)NQ * sizeof(float));
  float*  Vmean = (float*)carve((size_t)H * NQ * DH * sizeof(float));
  float*  Vsq   = (float*)carve((size_t)H * NQ * DH * sizeof(float));
  float*  muB   = (float*)carve((size_t)H * DH * sizeof(float));
  float*  invB  = (float*)carve((size_t)H * DH * sizeof(float));
  bf16_t* yb    = (bf16_t*)carve((size_t)NQ * QD * sizeof(bf16_t));

  const int TB = 256;
  auto nb = [](int n) { return (n + 255) / 256; };

  // 1) bf16 casts of x and weights
  f32_to_bf16<<<nb(NQ * QD), TB, 0, stream>>>(x, xb, NQ * QD);
  f32_to_bf16<<<nb(QD * QD), TB, 0, stream>>>(Wq, Wqb, QD * QD);
  f32_to_bf16<<<nb(QD * QD), TB, 0, stream>>>(Wk, Wkb, QD * QD);
  f32_to_bf16<<<nb(QD * QD), TB, 0, stream>>>(Wv, Wvb, QD * QD);
  f32_to_bf16<<<nb(QD * QD), TB, 0, stream>>>(Wo, Wob, QD * QD);

  // 2) projections: (2048x640) @ (640x640)^T -> f32 ; 5120 tiles / 8 waves
  dim3 gProj(640, 1);
  gemm_bf16_nt<<<gProj, TB, 0, stream>>>(xb, Wqb, qcs, NQ, QD, QD, QD, QD, QD,
                                         1.0f, nullptr);
  gemm_bf16_nt<<<gProj, TB, 0, stream>>>(xb, Wkb, kcs, NQ, QD, QD, QD, QD, QD,
                                         1.0f, nullptr);
  gemm_bf16_nt<<<gProj, TB, 0, stream>>>(xb, Wvb, vcs, NQ, QD, QD, QD, QD, QD,
                                         1.0f, nullptr);

  // 3) head packing (pad DH=80 -> DP=96)
  pack_q<<<nb(H * NQ * DP), TB, 0, stream>>>(qcs, Qc, qf, qb);
  pack_head_bf16<<<nb(H * NS * DP), TB, 0, stream>>>(Ks, Ksb, 0);
  pack_head_bf16<<<nb(H * NS * DP), TB, 0, stream>>>(kcs, kcsb, 1);
  pack_vt<<<nb(H * DH * NTOK), TB, 0, stream>>>(Vs, vcs, vtb, vsqtb);

  // 4) logits: 8-tile N-strips per wave; 2048 strips/head -> grid (256, H)
  dim3 gLog(256, H);
  gemm_logits<<<gLog, TB, 0, stream>>>(qb, Ksb, logit, 0, T_INJ * SCALE);
  gemm_logits<<<gLog, TB, 0, stream>>>(qb, kcsb, logit, NS, SCALE);

  // 5) per-row logsumexp gating -> lambda(n)
  row_lse<<<H * NQ, TB, 0, stream>>>(logit, rbuf);
  compute_lam<<<nb(NQ), TB, 0, stream>>>(rbuf, lamb);

  // 6) fused lambda-scale + top-k(128) + softmax (async LDS staging, in place)
  topk_softmax<<<H * NQ, TB, 0, stream>>>(logit, lamb);

  // 7) Vmean / Vsq: 16x80 row panel per wave -> attn matrix read once
  dim3 gAV(16, H);
  gemm_attn_v<<<gAV, TB, 0, stream>>>(logit, vtb, vsqtb, Vmean, Vsq);

  // 8) AdaIN modulation (out == Vmean, so out_final = Vmean + g*Vstd*Qn)
  q_stats<<<H * DH, TB, 0, stream>>>(qf, muB, invB);
  modulate<<<nb(H * NQ * DH), TB, 0, stream>>>(Vmean, Vsq, qf, muB, invB, gate, yb);

  // 9) output projection with bias -> d_out (f32)
  gemm_bf16_nt<<<gProj, TB, 0, stream>>>(yb, Wob, out, NQ, QD, QD, QD, QD, QD,
                                         1.0f, bo);
}
